// Attention_70909910057458
// MI455X (gfx1250) — matrix operand order repaired and checked by reading, not compiled, not run on hardware
//
#include <hip/hip_runtime.h>
#include <hip/hip_bf16.h>

// ---------------------------------------------------------------------------
// Flash attention for B=16, S=2048, D=256 (fp32 reference) on gfx1250.
//  - f32 -> bf16 one-pass conversion (Q,K row-major; V transposed)
//  - 4-wave workgroups stage K/V tiles into LDS with GLOBAL_LOAD_ASYNC_TO_LDS
//    (ASYNCcnt-tracked, double-buffered: prefetch tile i+1 during tile i)
//  - v_wmma_f32_16x16x32_bf16 for both GEMMs, f32 accum, f32 online softmax
//  - non-standard masking: masked scores forced to 0.0f (NOT -inf), still in
//    softmax; global mask.sum()>0 predicate gathered for free from the stream
// ---------------------------------------------------------------------------

#define B_ 16
#define S_ 2048
#define D_ 256
#define KT 32                 // k-tile columns per iteration
#define NITER (S_ / KT)       // 64

typedef __attribute__((ext_vector_type(16))) __bf16 v16bf;
typedef __attribute__((ext_vector_type(8)))  __bf16 v8bf;
typedef __attribute__((ext_vector_type(4)))  __bf16 v4bf;
typedef __attribute__((ext_vector_type(8)))  float  v8f;

// ---- fragment loaders -----------------------------------------------------
// 16-bit A-matrix 16x32, contiguous K order (this code's choice): lanes 0-15 hold M=lane with
// K = 0..15; lanes 16-31 hold M=lane-16 with K = 16..31 (matches B).
__device__ __forceinline__ v16bf make_afrag_bf16(const __bf16* rowp, int kOffA) {
    const v8bf r0 = *reinterpret_cast<const v8bf*>(rowp + kOffA);
    const v8bf r1 = *reinterpret_cast<const v8bf*>(rowp + 8 + kOffA);
    v16bf a;
#pragma unroll
    for (int e = 0; e < 8; ++e) { a[e] = r0[e]; a[e + 8] = r1[e]; }
    return a;
}

// 16-bit B-matrix 32x16 layout: lane n (0-15) holds column N=n with K=0..15
// contiguous; lane n+16 holds K=16..31. One 32B contiguous run per lane.
__device__ __forceinline__ v16bf load_bfrag(const __bf16* p) {
    return *reinterpret_cast<const v16bf*>(p);
}

// ---- phase 1: conversions --------------------------------------------------
__global__ void cvt_f32_to_bf16_kernel(const float* __restrict__ src,
                                       __bf16* __restrict__ dst, int n4) {
    int i = blockIdx.x * blockDim.x + threadIdx.x;
    if (i >= n4) return;
    const float4 f = reinterpret_cast<const float4*>(src)[i];
    v4bf o;
    o[0] = (__bf16)f.x; o[1] = (__bf16)f.y; o[2] = (__bf16)f.z; o[3] = (__bf16)f.w;
    reinterpret_cast<v4bf*>(dst)[i] = o;
}

// V[b,s,d] f32 -> VT[b,d,s] bf16 via 32x32 LDS tile (coalesced both sides).
__global__ void transpose_v_kernel(const float* __restrict__ v,
                                   __bf16* __restrict__ vt) {
    __shared__ __bf16 tile[32][33];
    const int b  = blockIdx.z;
    const int s0 = blockIdx.x * 32;
    const int d0 = blockIdx.y * 32;
    const int tx = threadIdx.x, ty = threadIdx.y;
#pragma unroll
    for (int r = 0; r < 4; ++r) {
        int srow = ty + r * 8;
        float f = v[((size_t)b * S_ + s0 + srow) * D_ + d0 + tx];
        tile[tx][srow] = (__bf16)f;
    }
    __syncthreads();
#pragma unroll
    for (int r = 0; r < 4; ++r) {
        int drow = ty + r * 8;
        vt[((size_t)b * D_ + d0 + drow) * S_ + s0 + tx] = tile[drow][tx];
    }
}

// ---- phase 2: flash attention ----------------------------------------------
// 4 waves per WG, each wave owns a 16-row q tile; WG shares async-staged K/V.
__global__ __launch_bounds__(128) void flash_attn_kernel(
    const __bf16* __restrict__ Qb, const __bf16* __restrict__ Kb,
    const __bf16* __restrict__ VT, const int* __restrict__ mask,
    float* __restrict__ out, unsigned* __restrict__ flag) {

    // [0,32K)  ldsK: 2 buffers of K tile   [KT][D_]  bf16 (16KB each)
    // [32K,64K) ldsV: 2 buffers of VT tile [D_][KT]  bf16 (16KB each)
    // P bounce (4KB) aliases the first 4KB of the *current* K buffer: K[cur]
    // is dead after the QK phase (barrier-separated), and the end-of-iteration
    // barrier retires the alias before the next prefetch overwrites it.
    __shared__ __align__(16) unsigned char ldsRaw[65536];
    __bf16* ldsK = (__bf16*)ldsRaw;
    __bf16* ldsV = (__bf16*)(ldsRaw + 32768);
    const uint32_t ldsKoff = (uint32_t)(uintptr_t)(void*)ldsRaw;        // LDS byte offset
    const uint32_t ldsVoff = ldsKoff + 32768;

    const int tid    = threadIdx.x;
    const int lane   = tid & 31;
    const int waveId = tid >> 5;
    const int laneLo = lane & 15;
    const int laneHi = lane >> 4;
    const int kOffA  = laneHi * 16;

    const int b     = blockIdx.x / (S_ / 64);
    const int q0    = (blockIdx.x % (S_ / 64)) * 64 + waveId * 16;

    const __bf16* Kbatch = Kb + (size_t)b * S_ * D_;
    const __bf16* Vbatch = VT + (size_t)b * D_ * S_;
    const int*    Mbase  = mask + ((size_t)b * S_ + q0) * S_;

    // Async-DMA one 32-row K tile (16KB) + one 32-col VT tile (16KB) into LDS
    // buffer `buf`. 1024 16B segments each, spread over 128 lanes: 16 async
    // instructions per wave per tile (ASYNCcnt in-order completion).
    auto issue_tile = [&](int k0, int buf) {
        const uint32_t kBase = ldsKoff + (uint32_t)buf * (KT * D_ * 2);
        const uint32_t vBase = ldsVoff + (uint32_t)buf * (D_ * KT * 2);
#pragma unroll
        for (int j = 0; j < 8; ++j) {
            const int s = j * 128 + tid;                       // K seg: row=s>>5, 32 segs/row
            const __bf16* gk = Kbatch + (size_t)(k0 + (s >> 5)) * D_ + (s & 31) * 8;
            asm volatile("global_load_async_to_lds_b128 %0, %1, off"
                         :: "v"(kBase + (uint32_t)s * 16), "v"(gk) : "memory");
        }
#pragma unroll
        for (int j = 0; j < 8; ++j) {
            const int s = j * 128 + tid;                       // V seg: row d=s>>2, 4 segs/row
            const __bf16* gv = Vbatch + (size_t)(s >> 2) * S_ + k0 + (s & 3) * 8;
            asm volatile("global_load_async_to_lds_b128 %0, %1, off"
                         :: "v"(vBase + (uint32_t)s * 16), "v"(gv) : "memory");
        }
    };

    // Q A-fragments: 8 chunks of K=32 covering D=256, resident all kernel.
    const __bf16* Qbase = Qb + ((size_t)b * S_ + q0 + laneLo) * D_;
    v16bf aq[8];
#pragma unroll
    for (int c = 0; c < 8; ++c) aq[c] = make_afrag_bf16(Qbase + c * 32, kOffA);

    v8f o[16];
    v8f vzero = {0.f, 0.f, 0.f, 0.f, 0.f, 0.f, 0.f, 0.f};
#pragma unroll
    for (int t = 0; t < 16; ++t) o[t] = vzero;

    float mrun[8], lrun[8];
#pragma unroll
    for (int j = 0; j < 8; ++j) { mrun[j] = -3.0e38f; lrun[j] = 0.0f; }
    int anyM = 0;

    issue_tile(0, 0);                                          // prologue prefetch

    for (int i = 0; i < NITER; ++i) {
        const int k0  = i * KT;
        const int cur = i & 1;

        if (i + 1 < NITER) {
            issue_tile(k0 + KT, cur ^ 1);                      // prefetch next tile
            asm volatile("s_wait_asynccnt 0x10" ::: "memory"); // drain tile i (in-order)
        } else {
            asm volatile("s_wait_asynccnt 0x0" ::: "memory");
        }
        __syncthreads();                                       // tile i visible WG-wide

        const __bf16* Kc = ldsK + cur * (KT * D_);
        const __bf16* Vc = ldsV + cur * (D_ * KT);
        __bf16*       pB = ldsK + cur * (KT * D_) + waveId * (16 * KT); // alias (see above)

        // ----- scores: S = Q * K^T (two 16x16 column halves, K-dim 256) -----
        v8f s0 = vzero, s1 = vzero;
#pragma unroll
        for (int c = 0; c < 8; ++c) {
            v16bf bk0 = load_bfrag(Kc + (size_t)laneLo * D_ + c * 32 + laneHi * 16);
            v16bf bk1 = load_bfrag(Kc + (size_t)(16 + laneLo) * D_ + c * 32 + laneHi * 16);
            s0 = __builtin_amdgcn_wmma_f32_16x16x32_bf16(false, aq[c], false, bk0,
                                                         (short)0, s0, false, false);
            s1 = __builtin_amdgcn_wmma_f32_16x16x32_bf16(false, aq[c], false, bk1,
                                                         (short)0, s1, false, false);
        }
        __syncthreads();   // all waves done reading K[cur] before P aliases it

        // ----- mask (masked -> 0.0f, still in softmax) + online softmax -----
#pragma unroll
        for (int j = 0; j < 8; ++j) {
            const int row = j + 8 * laneHi;                    // C/D: VGPR j = rows j, j+8
            const int m0 = Mbase[(size_t)row * S_ + k0 + laneLo];
            const int m1 = Mbase[(size_t)row * S_ + k0 + 16 + laneLo];
            anyM |= m0 | m1;
            float v0 = s0[j] * 0.0625f;                        // 1/sqrt(256)
            float v1 = s1[j] * 0.0625f;
            v0 = m0 ? v0 : 0.0f;
            v1 = m1 ? v1 : 0.0f;

            float tmax = fmaxf(v0, v1);
#pragma unroll
            for (int d = 1; d < 16; d <<= 1) tmax = fmaxf(tmax, __shfl_xor(tmax, d, 32));
            const float mnew = fmaxf(mrun[j], tmax);
            const float sc   = __expf(mrun[j] - mnew);
            const float p0   = __expf(v0 - mnew);
            const float p1   = __expf(v1 - mnew);
            float rsum = p0 + p1;
#pragma unroll
            for (int d = 1; d < 16; d <<= 1) rsum += __shfl_xor(rsum, d, 32);
            lrun[j] = lrun[j] * sc + rsum;
            mrun[j] = mnew;
#pragma unroll
            for (int t = 0; t < 16; ++t) o[t][j] *= sc;        // rescale accumulators
            pB[row * KT + laneLo]      = (__bf16)p0;           // stash P (bf16, row-major)
            pB[row * KT + 16 + laneLo] = (__bf16)p1;
        }
        asm volatile("s_wait_dscnt 0" ::: "memory");           // own P slice visible

        // ----- O += P * V  (K-dim = 32, 16 output d-tiles) -----
        const v16bf ap = make_afrag_bf16(pB + laneLo * KT, kOffA);
#pragma unroll
        for (int t = 0; t < 16; ++t) {
            v16bf bv = load_bfrag(Vc + (size_t)(t * 16 + laneLo) * KT + laneHi * 16);
            o[t] = __builtin_amdgcn_wmma_f32_16x16x32_bf16(false, ap, false, bv,
                                                           (short)0, o[t], false, false);
        }
        __syncthreads();   // retire buf[cur] (V + P alias) before next prefetch
    }

    // global mask.sum()>0 predicate: one atomic per wave, only if nonzero seen
    if (__any(anyM != 0)) { if (lane == 0) atomicOr(flag, 1u); }

    float rinv[8];
#pragma unroll
    for (int j = 0; j < 8; ++j) rinv[j] = 1.0f / lrun[j];
    float* obase = out + ((size_t)b * S_ + q0) * D_;
#pragma unroll
    for (int t = 0; t < 16; ++t)
#pragma unroll
        for (int j = 0; j < 8; ++j) {
            const int row = j + 8 * laneHi;
            obase[(size_t)row * D_ + t * 16 + laneLo] = o[t][j] * rinv[j];
        }
}

// ---- phase 3: zero output iff mask.sum()==0 (reference edge case) -----------
__global__ void fixup_kernel(float* __restrict__ out,
                             const unsigned* __restrict__ flag, int n4) {
    if (*flag != 0u) return;                  // ~always: one cached scalar read
    int i = blockIdx.x * blockDim.x + threadIdx.x;
    if (i >= n4) return;
    float4 z = {0.f, 0.f, 0.f, 0.f};
    reinterpret_cast<float4*>(out)[i] = z;
}

// ---------------------------------------------------------------------------
extern "C" void kernel_launch(void* const* d_in, const int* in_sizes, int n_in,
                              void* d_out, int out_size, void* d_ws, size_t ws_size,
                              hipStream_t stream) {
    (void)in_sizes; (void)n_in; (void)out_size; (void)ws_size;
    const float* q    = (const float*)d_in[0];
    const float* k    = (const float*)d_in[1];
    const float* v    = (const float*)d_in[2];
    const int*   mask = (const int*)d_in[3];
    float* out = (float*)d_out;

    const size_t nQ = (size_t)B_ * S_ * D_;          // 8,388,608 elements
    __bf16* Qb = (__bf16*)d_ws;                      // 16.8 MB
    __bf16* Kb = Qb + nQ;                            // 16.8 MB
    __bf16* VT = Kb + nQ;                            // 16.8 MB (transposed V)
    unsigned* flag = (unsigned*)(VT + nQ);           // 4 B mask-nonzero flag

    const int n4 = (int)(nQ / 4);
    cvt_f32_to_bf16_kernel<<<(n4 + 255) / 256, 256, 0, stream>>>(q, Qb, n4);
    cvt_f32_to_bf16_kernel<<<(n4 + 255) / 256, 256, 0, stream>>>(k, Kb, n4);
    transpose_v_kernel<<<dim3(S_ / 32, D_ / 32, B_), dim3(32, 8), 0, stream>>>(v, VT);
    hipMemsetAsync(flag, 0, sizeof(unsigned), stream);

    flash_attn_kernel<<<B_ * (S_ / 64), 128, 0, stream>>>(Qb, Kb, VT, mask, out, flag);

    fixup_kernel<<<(n4 + 255) / 256, 256, 0, stream>>>(out, flag, n4);
}
